// LinearAWQ_44976897524097
// MI455X (gfx1250) — matrix operand order repaired and checked by reading, not compiled, not run on hardware
//
#include <hip/hip_runtime.h>

// AWQ W4A16 GEMM for gfx1250 (MI455X): x[4096,3584]f16 @ dequant(W)[3584,18944] -> f16
// Compute-bound -> maximize v_wmma_f32_16x16x32_f16 density:
//  - double-buffered LDS, one barrier per K-step
//  - x tile staged via global_load_async_to_lds_b128 (ASYNCcnt)
//  - dequant via 0x6400|nibble fp16 bit trick + v_pk_fma_f16 (no int->float cvt)

typedef _Float16 half_t;
typedef __attribute__((ext_vector_type(16))) _Float16 v16h;
typedef __attribute__((ext_vector_type(8)))  _Float16 v8h;
typedef __attribute__((ext_vector_type(2)))  _Float16 h2;
typedef __attribute__((ext_vector_type(8)))  float    v8f;

#define M_DIM 4096
#define K_DIM 3584
#define N_DIM 18944
#define NP    (N_DIM / 8)     // 2368 packed int32 columns
#define GROUP 128
#define BM 128
#define BN 128
#define BK 32
#define LDA 40                // halves per As row  (80B stride, 16B aligned)
#define LDB 40                // halves per Bs[n] row (K-contiguous)
#define KSTEPS (K_DIM / BK)   // 112

__global__ __launch_bounds__(256)
void awq_gemm_wmma(const half_t* __restrict__ x,
                   const int*    __restrict__ qweight,
                   const int*    __restrict__ qzeros,
                   const half_t* __restrict__ scales,
                   half_t*       __restrict__ out)
{
    __shared__ half_t As[2][BM * LDA];   // x tile, [m][k]
    __shared__ half_t Bs[2][BN * LDB];   // dequant W tile, [n][k]

    const int tid  = threadIdx.x;
    const int lane = tid & 31;
    const int wv   = tid >> 5;        // 8 waves
    const int wrow = wv >> 2;         // 0..1 -> 64 M-rows per wave
    const int wcol = wv & 3;          // 0..3 -> 32 N-cols per wave

    const int m0  = blockIdx.y * BM;
    const int n0  = blockIdx.x * BN;
    const int nb0 = n0 >> 3;

    // x staging: thread owns row tid>>1, 16 halves (32B) at xcol
    const int xrow = tid >> 1;
    const int xcol = (tid & 1) * 16;
    const half_t* xp = x + (size_t)(m0 + xrow) * K_DIM + xcol;

    // W staging: thread owns packed col ci at k-rows (kr2, kr2+1)
    const int kr2 = (tid >> 4) * 2;   // 0,2,...,30
    const int ci  = tid & 15;         // 0..15

    // AWQ interleave: nibble for output col 8c+j at shift 4*[0,4,1,5,2,6,3,7][j]
    const int sh[8] = {0, 16, 4, 20, 8, 24, 12, 28};

    // Per-group dequant tables (renewed every 4 K-steps):
    //   svd[j] = {s,s},  zb2[j] = -(1024+z)*{s,s}
    // so dequant of a nibble pair is one v_pk_fma_f16 on (0x6400|nib) bits.
    h2 svd[8], zb2[8];

    v8f acc[4][2];
    #pragma unroll
    for (int i = 0; i < 4; ++i)
        #pragma unroll
        for (int j = 0; j < 2; ++j)
            acc[i][j] = (v8f){0.f, 0.f, 0.f, 0.f, 0.f, 0.f, 0.f, 0.f};

    const int mrow = lane & 15;
    const int kbA  = (lane & 16) ? 8  : 0;
    const int kbB  = (lane & 16) ? 8  : 0;

    auto load_group = [&](int g) {
        const int z32 = qzeros[(size_t)g * NP + nb0 + ci];
        const v8h sv  = *(const v8h*)(scales + (size_t)g * N_DIM + n0 + ci * 8);
        #pragma unroll
        for (int j = 0; j < 8; ++j) {
            const half_t s = sv[j];
            h2 sd; sd[0] = s; sd[1] = s;
            const unsigned short zb =
                (unsigned short)(0x6400u | ((unsigned)(z32 >> sh[j]) & 0xFu));
            const half_t zf = __builtin_bit_cast(_Float16, zb);
            h2 zd; zd[0] = zf; zd[1] = zf;
            svd[j] = sd;
            zb2[j] = -(zd * sd);
        }
    };

    auto stage = [&](int t, int p) {
        const int k0 = t * BK;
        // x tile: async DMA global->LDS, 2 x 16B per thread (ASYNCcnt).
        {
            const unsigned long long ga = (unsigned long long)(size_t)(xp + k0);
            const unsigned lds0 = (unsigned)(size_t)&As[p][xrow * LDA + xcol];
            asm volatile("global_load_async_to_lds_b128 %0, %1, off"
                         :: "v"(lds0), "v"(ga) : "memory");
            asm volatile("global_load_async_to_lds_b128 %0, %1, off offset:16"
                         :: "v"(lds0), "v"(ga) : "memory");
        }
        // W tile: two k-adjacent packed dwords of this thread's column.
        const unsigned qa = (unsigned)qweight[(size_t)(k0 + kr2)     * NP + nb0 + ci];
        const unsigned qb = (unsigned)qweight[(size_t)(k0 + kr2 + 1) * NP + nb0 + ci];
        half_t* bp = &Bs[p][(ci * 8) * LDB + kr2];
        #pragma unroll
        for (int j = 0; j < 8; ++j) {
            const unsigned nib2 = ((qa >> sh[j]) & 0xFu) | (((qb >> sh[j]) & 0xFu) << 16);
            const h2 w = __builtin_bit_cast(h2, nib2 | 0x64006400u);  // {1024+wk, 1024+wk1}
            const h2 v = w * svd[j] + zb2[j];                         // v_pk_fma_f16
            *(h2*)(bp + j * LDB) = v;                                 // ds_store_b32
        }
    };

    load_group(0);
    stage(0, 0);
    asm volatile("s_wait_asynccnt 0" ::: "memory");
    __syncthreads();

    for (int t = 0; t < KSTEPS; ++t) {
        const int p = t & 1;
        if (t + 1 < KSTEPS) {
            if (((t + 1) & 3) == 0) load_group((t + 1) >> 2);
            stage(t + 1, p ^ 1);
        }

        // A fragments: 16-bit A 16x32 layout (two contiguous 16B LDS reads/lane)
        v16h a[4];
        #pragma unroll
        for (int i = 0; i < 4; ++i) {
            const half_t* pa = &As[p][(wrow * 64 + i * 16 + mrow) * LDA + kbA];
            const v8h lo = *(const v8h*)pa;
            const v8h hi = *(const v8h*)(pa + 16);
            v16h av;
            #pragma unroll
            for (int e = 0; e < 8; ++e) { av[e] = lo[e]; av[8 + e] = hi[e]; }
            a[i] = av;
        }
        // B fragments: 16-bit B 32x16 layout, K-contiguous via Bs[n][k]
        v16h b[2];
        #pragma unroll
        for (int j = 0; j < 2; ++j) {
            const half_t* pb = &Bs[p][(wcol * 32 + j * 16 + mrow) * LDB + kbB];
            const v8h lo = *(const v8h*)pb;
            const v8h hi = *(const v8h*)(pb + 16);
            v16h bv;
            #pragma unroll
            for (int e = 0; e < 8; ++e) { bv[e] = lo[e]; bv[8 + e] = hi[e]; }
            b[j] = bv;
        }

        #pragma unroll
        for (int i = 0; i < 4; ++i)
            #pragma unroll
            for (int j = 0; j < 2; ++j)
                acc[i][j] = __builtin_amdgcn_wmma_f32_16x16x32_f16(
                    false, a[i], false, b[j], (short)0, acc[i][j], false, false);

        asm volatile("s_wait_asynccnt 0" ::: "memory");
        __syncthreads();
    }

    // Epilogue: C/D layout — VGPR r: lanes 0-15 -> M=r, lanes 16-31 -> M=r+8
    #pragma unroll
    for (int i = 0; i < 4; ++i) {
        #pragma unroll
        for (int j = 0; j < 2; ++j) {
            const int col  = n0 + wcol * 32 + j * 16 + mrow;
            const int row0 = m0 + wrow * 64 + i * 16 + ((lane & 16) ? 8 : 0);
            #pragma unroll
            for (int r = 0; r < 8; ++r)
                out[(size_t)(row0 + r) * N_DIM + col] = (half_t)acc[i][j][r];
        }
    }
}

extern "C" void kernel_launch(void* const* d_in, const int* in_sizes, int n_in,
                              void* d_out, int out_size, void* d_ws, size_t ws_size,
                              hipStream_t stream) {
    (void)in_sizes; (void)n_in; (void)out_size; (void)d_ws; (void)ws_size;
    const half_t* xp = (const half_t*)d_in[0];
    const int*    qw = (const int*)d_in[1];
    const int*    qz = (const int*)d_in[2];
    const half_t* sc = (const half_t*)d_in[3];
    half_t*       op = (half_t*)d_out;
    dim3 grid(N_DIM / BN, M_DIM / BM);   // 148 x 32
    awq_gemm_wmma<<<grid, 256, 0, stream>>>(xp, qw, qz, sc, op);
}
